// ComputeAttention_21552145891838
// MI455X (gfx1250) — compile-verified, matrix operand order checked by reading, not run on hardware
//
#include <hip/hip_runtime.h>
#include <hip/hip_bf16.h>

#define B_    8
#define T_EN  512
#define T_DE  128
#define D_EN  512
#define D_DE  512
#define UNITS 256

typedef __attribute__((ext_vector_type(16))) __bf16 v16bf;
typedef __attribute__((ext_vector_type(8)))  float  v8f;

// ---------------------------------------------------------------------------
// fast tanh: prefer native v_tanh_f32 (gfx1250), fall back to v_exp_f32 form
// ---------------------------------------------------------------------------
__device__ __forceinline__ float fast_exp2(float x) {
#if __has_builtin(__builtin_amdgcn_exp2f)
    return __builtin_amdgcn_exp2f(x);
#else
    return exp2f(x);
#endif
}

__device__ __forceinline__ float fast_tanh(float x) {
#if __has_builtin(__builtin_amdgcn_tanhf)
    return __builtin_amdgcn_tanhf(x);
#elif __has_builtin(__builtin_amdgcn_tanh_f32)
    return __builtin_amdgcn_tanh_f32(x);
#else
    float e = fast_exp2(x * 2.8853900817779268f);   // exp(2x)
    return 1.0f - 2.0f / (e + 1.0f);
#endif
}

__device__ __forceinline__ v8f wmma_bf16(const v16bf& a, const v16bf& b, const v8f& c) {
    return __builtin_amdgcn_wmma_f32_16x16x32_bf16(
        /*neg_a=*/false, a, /*neg_b=*/false, b,
        /*c_mod=*/(short)0, c, /*reuse_a=*/false, /*reuse_b=*/false);
}

// ---------------------------------------------------------------------------
// Stage 1: C[b, M, N] = A[b, M, K] x W[K, N]   (f32 in, bf16 WMMA, f32 out)
// One wave computes a 32x32 C tile: 2 A-frags x 2 B-frags -> 4 WMMAs per
// 32-deep k-step (2x operand reuse vs a 16x16 tile).
// Fragment packing per CDNA5 ISA 7.12.2 (16-bit A 16x32 / B 32x16):
//   lane&15 = row (A) / col (B); lane>>4 selects K-half (+8);
//   VGPR vg holds K pair kp = (vg&3)*2 + (vg>>2)*16 + khalf
// C/D 16x16 f32: VGPR r -> M = r + (lane>=16 ? 8 : 0), N = lane&15
// M,K,N compile-time so all addressing strength-reduces to shifts/adds.
// ---------------------------------------------------------------------------
template <int M, int K, int N>
__global__ void wmma_gemm_bf16_32x32_kernel(const float* __restrict__ A,
                                            const float* __restrict__ W,
                                            float* __restrict__ C) {
    const int lane   = threadIdx.x & 31;
    const int waveId = (blockIdx.x * (blockDim.x >> 5)) + (threadIdx.x >> 5);

    constexpr int tilesN = N >> 5;
    constexpr int tilesM = M >> 5;
    constexpr int tilesPerBatch = tilesM * tilesN;

    const int b  = waveId / tilesPerBatch;
    const int t  = waveId % tilesPerBatch;
    const int m0 = (t / tilesN) << 5;
    const int n0 = (t % tilesN) << 5;

    const float* Ab = A + (size_t)b * M * K;
    float*       Cb = C + (size_t)b * M * N;

    const int hidx  = lane & 15;          // row (A frags) / col (B frags)
    const int khalf = (lane >> 4) << 3;   // 0 or 8

    v8f acc00 = {}, acc01 = {}, acc10 = {}, acc11 = {};

#pragma unroll 2
    for (int k0 = 0; k0 < K; k0 += 32) {
        v16bf a0, a1, b0, b1;
#pragma unroll
        for (int vg = 0; vg < 8; ++vg) {
            const int kp = ((vg & 3) << 1) + ((vg >> 2) << 4) + khalf;
            // A rows m0+hidx and m0+16+hidx, consecutive K pair
            const float* ar0 = Ab + (size_t)(m0 + hidx)      * K + (k0 + kp);
            const float* ar1 = Ab + (size_t)(m0 + 16 + hidx) * K + (k0 + kp);
            a0[2 * vg]     = (__bf16)ar0[0];
            a0[2 * vg + 1] = (__bf16)ar0[1];
            a1[2 * vg]     = (__bf16)ar1[0];
            a1[2 * vg + 1] = (__bf16)ar1[1];
            // W rows k0+kp, k0+kp+1; columns n0+hidx and n0+16+hidx
            const float* wr0 = W + (size_t)(k0 + kp)     * N + (n0 + hidx);
            const float* wr1 = W + (size_t)(k0 + kp + 1) * N + (n0 + hidx);
            b0[2 * vg]     = (__bf16)wr0[0];
            b0[2 * vg + 1] = (__bf16)wr1[0];
            b1[2 * vg]     = (__bf16)wr0[16];
            b1[2 * vg + 1] = (__bf16)wr1[16];
        }
        acc00 = wmma_bf16(a0, b0, acc00);
        acc01 = wmma_bf16(a0, b1, acc01);
        acc10 = wmma_bf16(a1, b0, acc10);
        acc11 = wmma_bf16(a1, b1, acc11);
    }

    const int col   = lane & 15;
    const int rbase = (lane >> 4) << 3;
#pragma unroll
    for (int r = 0; r < 8; ++r) {
        const int row0 = m0 + rbase + r;
        Cb[(size_t)row0 * N + (n0 + col)]        = acc00[r];
        Cb[(size_t)row0 * N + (n0 + 16 + col)]   = acc01[r];
        const int row1 = row0 + 16;
        Cb[(size_t)row1 * N + (n0 + col)]        = acc10[r];
        Cb[(size_t)row1 * N + (n0 + 16 + col)]   = acc11[r];
    }
}

// ---------------------------------------------------------------------------
// Stage 2: fused  mu[b,j,i] = sum_u nu[u]*tanh(attDe[b,j,u]+attEn[b,i,u]),
//          mask, softmax over i.  One block (256 threads, 8 waves) per (b,j).
// Wave w handles rows i = w, w+8, ...; lanes split U=256 into 8 chunks of 32
// (coalesced, L2-resident).  Transcendental-VALU bound: ~134M v_tanh total.
// ---------------------------------------------------------------------------
__global__ void tanh_dot_softmax_kernel(const float* __restrict__ attEn,
                                        const float* __restrict__ attDe,
                                        const float* __restrict__ nu,
                                        const int*   __restrict__ maskEn,
                                        float* __restrict__ out) {
    const int bj   = blockIdx.x;
    const int b    = bj / T_DE;
    const int j    = bj % T_DE;
    const int tid  = threadIdx.x;
    const int lane = tid & 31;
    const int wave = tid >> 5;

    __shared__ float mu[T_EN];
    __shared__ float red[256];

    const float* de = attDe + ((size_t)b * T_DE + j) * UNITS;
    const float* en = attEn + (size_t)b * T_EN * UNITS;
    const int*   mr = maskEn + (size_t)b * T_EN;

    float dereg[8], nureg[8];
#pragma unroll
    for (int c = 0; c < 8; ++c) {
        dereg[c] = de[lane + 32 * c];
        nureg[c] = nu[lane + 32 * c];
    }

    for (int i = wave; i < T_EN; i += 8) {
        const float* enr = en + (size_t)i * UNITS;
        float p = 0.0f;
#pragma unroll
        for (int c = 0; c < 8; ++c) {
            p += nureg[c] * fast_tanh(dereg[c] + enr[lane + 32 * c]);
        }
#pragma unroll
        for (int off = 16; off >= 1; off >>= 1)
            p += __shfl_xor(p, off, 32);
        if (lane == 0) {
            const float m = (float)mr[i];
            mu[i] = p + (m - 1.0f) * 1000000.0f;   // mask: -1e6 where m==0
        }
    }
    __syncthreads();

    // block max over 512 logits
    red[tid] = fmaxf(mu[tid], mu[tid + 256]);
    __syncthreads();
    for (int s = 128; s >= 1; s >>= 1) {
        if (tid < s) red[tid] = fmaxf(red[tid], red[tid + s]);
        __syncthreads();
    }
    const float mx = red[0];
    __syncthreads();

    // exp + block sum
    const float e0 = fast_exp2((mu[tid]       - mx) * 1.4426950408889634f);
    const float e1 = fast_exp2((mu[tid + 256] - mx) * 1.4426950408889634f);
    red[tid] = e0 + e1;
    __syncthreads();
    for (int s = 128; s >= 1; s >>= 1) {
        if (tid < s) red[tid] += red[tid + s];
        __syncthreads();
    }
    const float inv = 1.0f / red[0];

    float* orow = out + ((size_t)b * T_DE + j) * T_EN;
    orow[tid]       = e0 * inv;
    orow[tid + 256] = e1 * inv;
}

// ---------------------------------------------------------------------------
extern "C" void kernel_launch(void* const* d_in, const int* in_sizes, int n_in,
                              void* d_out, int out_size, void* d_ws, size_t ws_size,
                              hipStream_t stream) {
    (void)in_sizes; (void)n_in; (void)out_size; (void)ws_size;

    const float* en_seq  = (const float*)d_in[0];   // [B, T_EN, D_EN]
    const float* de_seq  = (const float*)d_in[1];   // [B, T_DE, D_DE]
    const int*   mask_en = (const int*)  d_in[2];   // [B, T_EN]
    const float* w_en    = (const float*)d_in[3];   // [D_EN, UNITS]
    const float* w_de    = (const float*)d_in[4];   // [D_DE, UNITS]
    const float* nu      = (const float*)d_in[5];   // [UNITS, 1]
    float*       out     = (float*)d_out;           // [B, T_DE, T_EN]

    float* attEn = (float*)d_ws;                              // B*T_EN*UNITS f32
    float* attDe = attEn + (size_t)B_ * T_EN * UNITS;         // B*T_DE*UNITS f32

    // GEMM 1: att_en  -> 8 * 16 * 8 = 1024 waves (32x32 tiles), 8 waves/block
    {
        const int waves  = B_ * (T_EN / 32) * (UNITS / 32);
        wmma_gemm_bf16_32x32_kernel<T_EN, D_EN, UNITS>
            <<<waves / 8, 256, 0, stream>>>(en_seq, w_en, attEn);
    }
    // GEMM 2: att_de  -> 8 * 4 * 8 = 256 waves
    {
        const int waves  = B_ * (T_DE / 32) * (UNITS / 32);
        wmma_gemm_bf16_32x32_kernel<T_DE, D_DE, UNITS>
            <<<waves / 8, 256, 0, stream>>>(de_seq, w_de, attDe);
    }
    // Fused tanh-dot + mask + softmax: one block per (b, j)
    tanh_dot_softmax_kernel<<<B_ * T_DE, 256, 0, stream>>>(attEn, attDe, nu,
                                                           mask_en, out);
}
